// MixtureOfExperts_74079595922110
// MI455X (gfx1250) — matrix operand order repaired and checked by reading, not compiled, not run on hardware
//
#include <hip/hip_runtime.h>
#include <hip/hip_bf16.h>
#include <math.h>

typedef __attribute__((ext_vector_type(16))) _Float16 v16h;
typedef __attribute__((ext_vector_type(8)))  float    v8f;
typedef __attribute__((ext_vector_type(4)))  int      v4i;

#define DD 128    // feature dim
#define HH 256    // hidden dim
#define EE 8      // experts
#define TPB 256   // tokens per block (2 M-tiles of 128)
#define XS_STRIDE 136   // halves; 272B row stride -> conflict-free, 16B aligned
#define WS_STRIDE 136

#define AS1 __attribute__((address_space(1)))
#define AS3 __attribute__((address_space(3)))

#if defined(__has_builtin)
#if __has_builtin(__builtin_amdgcn_global_load_async_to_lds_b128)
#define HAVE_ASYNC_LDS 1
#endif
#endif

// ---- prep: W1 [E][D][H] f32  ->  w1t [E][H][D] f16 (contiguous) ----
__global__ __launch_bounds__(256) void w1_to_f16t_kernel(
    const float* __restrict__ W1, _Float16* __restrict__ w1t)
{
    int idx = blockIdx.x * 256 + threadIdx.x;   // over E*H*D = 262144
    int e = idx >> 15;
    int r = idx & 32767;
    int h = r >> 7;
    int d = r & 127;
    w1t[idx] = (_Float16)W1[(((size_t)e << 7) + d) * HH + h];
}

__global__ __launch_bounds__(256) void moe_wmma_kernel(
    const float* __restrict__ x,
    const float* __restrict__ cp,
    const float* __restrict__ W1,
    const float* __restrict__ b1,
    const float* __restrict__ W2,
    const float* __restrict__ b2,
    const _Float16* __restrict__ w1t,   // may be null -> in-kernel conversion
    float* __restrict__ pred,
    float* __restrict__ gates,
    float* __restrict__ experts_used,
    float* __restrict__ infl,
    float* __restrict__ cnt)
{
    __shared__ __align__(16) _Float16 xs[TPB * XS_STRIDE];   // x tile [tok][d] f16
    __shared__ __align__(16) _Float16 w1s[HH * WS_STRIDE];   // W1[e]^T [h][d] f16
    __shared__ float b1s[HH];
    __shared__ float w2s[HH];
    __shared__ float gsm[TPB * EE];
    __shared__ float pred_s[TPB];

    const int t    = threadIdx.x;
    const int lane = t & 31;
    const int wave = t >> 5;           // 0..7 -> M-subtile within each 128-token tile
    const int tok0 = blockIdx.x * TPB;

    // ---- stage x tile to LDS as f16 (float2 -> packed half2 stores) ----
    #pragma unroll 8
    for (int i = 0; i < (TPB * DD) / 512; ++i) {             // 64 iters
        int j   = (i * 256 + t) * 2;
        int tok = j / DD, d = j % DD;
        float2 v = *(const float2*)(x + (size_t)(tok0 + tok) * DD + d);
        union { _Float16 h[2]; unsigned u; } cvt;
        cvt.h[0] = (_Float16)v.x;  cvt.h[1] = (_Float16)v.y;
        *(unsigned*)&xs[tok * XS_STRIDE + d] = cvt.u;
    }

    // ---- softmax gates, experts_used (one token per thread) ----
    {
        float g[EE];
        const float* cpt = cp + (size_t)(tok0 + t) * EE;
        float mx = -3.4e38f;
        #pragma unroll
        for (int e = 0; e < EE; ++e) { g[e] = cpt[e]; mx = fmaxf(mx, g[e]); }
        float s = 0.f;
        #pragma unroll
        for (int e = 0; e < EE; ++e) { g[e] = __expf(g[e] - mx); s += g[e]; }
        float inv = 1.f / s, used = 0.f;
        #pragma unroll
        for (int e = 0; e < EE; ++e) {
            float gv = g[e] * inv;
            gsm[t * EE + e] = gv;
            gates[(size_t)(tok0 + t) * EE + e] = gv;
            used += (gv > 0.01f) ? 1.f : 0.f;
        }
        experts_used[tok0 + t] = used;
        pred_s[t] = 0.f;
    }
    __syncthreads();

    // ---- block-level expert influence / activation count ----
    if (t < EE) {
        float si = 0.f, sc = 0.f;
        for (int i = 0; i < TPB; ++i) {
            float gv = gsm[i * EE + t];
            si += gv;
            sc += (gv > 0.01f) ? 1.f : 0.f;
        }
        atomicAdd(&infl[t], si);
        atomicAdd(&cnt[t], sc);
    }

    // ---- A fragments (x tile), invariant across experts ----
    // 16x32 f16 A layout: lane L holds row M=L%16; elems 0..7 -> K=g*8+0..7,
    // elems 8..15 -> K=16+g*8+0..7, g=L>>4.
    const int g      = lane >> 4;
    const int n_lane = lane & 15;
    v16h afrag[2][4];
    #pragma unroll
    for (int tt = 0; tt < 2; ++tt) {
        const int mrow = tt * 128 + (wave << 4) + n_lane;
        #pragma unroll
        for (int kb = 0; kb < 4; ++kb) {
            const _Float16* arow = &xs[mrow * XS_STRIDE + kb * 32 + g * 8];
            uint4* ap = (uint4*)&afrag[tt][kb];
            ap[0] = *(const uint4*)(arow);
            ap[1] = *(const uint4*)(arow + 16);
        }
    }

    // ---- expert loop ----
    for (int e = 0; e < EE; ++e) {
        __syncthreads();   // previous compute done before w1s overwrite

        if (w1t) {
            // pre-converted f16 W1^T: pure copy, padded LDS layout kept in dest addrs
            const _Float16* w1base = w1t + (size_t)e * (HH * DD);
            #pragma unroll 4
            for (int i = 0; i < (HH * DD) / (256 * 8); ++i) {   // 16 chunks/thread
                int c = i * 256 + t;
                int j = c * 8;                  // half index in [H][D]
                int h = j >> 7, d = j & 127;
                const _Float16* gp = w1base + j;
                _Float16* lp = &w1s[h * WS_STRIDE + d];
#ifdef HAVE_ASYNC_LDS
                __builtin_amdgcn_global_load_async_to_lds_b128(
                    (AS1 v4i*)gp, (AS3 v4i*)lp, 0, 0);
#else
                *(uint4*)lp = *(const uint4*)gp;
#endif
            }
#ifdef HAVE_ASYNC_LDS
#if defined(__has_builtin) && __has_builtin(__builtin_amdgcn_s_wait_asynccnt)
            __builtin_amdgcn_s_wait_asynccnt(0);
#else
            asm volatile("s_wait_asynccnt 0x0" ::: "memory");
#endif
#endif
        } else {
            // fallback: convert+transpose f32 W1[e] in-kernel
            const float* w1e = W1 + (size_t)e * DD * HH;
            #pragma unroll 4
            for (int i = 0; i < (DD * HH) / 256; ++i) {          // 128 iters
                int j = i * 256 + t;
                int d = j >> 8, h = j & 255;                     // HH == 256
                w1s[h * WS_STRIDE + d] = (_Float16)w1e[j];
            }
        }
        b1s[t] = b1[e * HH + t];
        w2s[t] = W2[(size_t)e * HH + t];
        __syncthreads();

        const float b2e = b2[e];
        float p0[8] = {0.f,0.f,0.f,0.f,0.f,0.f,0.f,0.f};
        float p1[8] = {0.f,0.f,0.f,0.f,0.f,0.f,0.f,0.f};

        for (int nt = 0; nt < 16; ++nt) {
            const int n = nt * 16 + n_lane;
            // 32x16 f16 B layout: lane L holds col N=L%16; elems 0..7 -> K=g*8+0..7, elems 8..15 -> K=16+g*8+0..7
            const _Float16* brow = &w1s[n * WS_STRIDE];
            v8f a0 = {0.f,0.f,0.f,0.f,0.f,0.f,0.f,0.f};
            v8f a1 = {0.f,0.f,0.f,0.f,0.f,0.f,0.f,0.f};
            #pragma unroll
            for (int kb = 0; kb < 4; ++kb) {
                v16h bfrag;
                uint4* bp = (uint4*)&bfrag;
                const _Float16* bk = brow + kb * 32 + g * 8;
                bp[0] = *(const uint4*)(bk);
                bp[1] = *(const uint4*)(bk + 16);
                a0 = __builtin_amdgcn_wmma_f32_16x16x32_f16(
                        false, afrag[0][kb], false, bfrag, (short)0, a0, false, false);
                a1 = __builtin_amdgcn_wmma_f32_16x16x32_f16(
                        false, afrag[1][kb], false, bfrag, (short)0, a1, false, false);
            }
            const float b1v = b1s[n], w2v = w2s[n];
            #pragma unroll
            for (int r = 0; r < 8; ++r) {
                float h0 = fmaxf(a0[r] + b1v, 0.f);  p0[r] += h0 * w2v;
                float h1 = fmaxf(a1[r] + b1v, 0.f);  p1[r] += h1 * w2v;
            }
        }

        // reduce partials across the 16 lanes of each D-matrix half-group
        #pragma unroll
        for (int r = 0; r < 8; ++r) {
            float v0 = p0[r], v1 = p1[r];
            v0 += __shfl_xor(v0, 1); v1 += __shfl_xor(v1, 1);
            v0 += __shfl_xor(v0, 2); v1 += __shfl_xor(v1, 2);
            v0 += __shfl_xor(v0, 4); v1 += __shfl_xor(v1, 4);
            v0 += __shfl_xor(v0, 8); v1 += __shfl_xor(v1, 8);
            if (n_lane == 0) {
                int tokl = (wave << 4) + (g << 3) + r;      // token in tile 0
                float y0 = 1.f / (1.f + __expf(-(v0 + b2e)));
                pred_s[tokl] += y0 * gsm[tokl * EE + e];
                int tokh = 128 + tokl;                       // token in tile 1
                float y1 = 1.f / (1.f + __expf(-(v1 + b2e)));
                pred_s[tokh] += y1 * gsm[tokh * EE + e];
            }
        }
    }

    __syncthreads();
    pred[tok0 + t] = pred_s[t];
}

extern "C" void kernel_launch(void* const* d_in, const int* in_sizes, int n_in,
                              void* d_out, int out_size, void* d_ws, size_t ws_size,
                              hipStream_t stream) {
    const float* x  = (const float*)d_in[0];
    const float* cp = (const float*)d_in[1];
    const float* W1 = (const float*)d_in[2];
    const float* b1 = (const float*)d_in[3];
    const float* W2 = (const float*)d_in[4];
    const float* b2 = (const float*)d_in[5];

    const int B = in_sizes[0] / DD;   // 131072

    float* out  = (float*)d_out;
    float* pred = out;                                // [B]
    float* gts  = out + (size_t)B;                    // [B,E]
    float* eu   = out + (size_t)B + (size_t)B * EE;   // [B]
    float* infl = eu + (size_t)B;                     // [E]
    float* cnt  = infl + EE;                          // [E]

    // zero the atomic accumulators (capturable memset node)
    (void)hipMemsetAsync(infl, 0, 2 * EE * sizeof(float), stream);

    const size_t w1t_bytes = (size_t)EE * HH * DD * sizeof(_Float16);  // 512 KB
    _Float16* w1t = (ws_size >= w1t_bytes) ? (_Float16*)d_ws : nullptr;
    if (w1t) {
        hipLaunchKernelGGL(w1_to_f16t_kernel, dim3((EE * HH * DD) / 256), dim3(256),
                           0, stream, W1, w1t);
    }

    dim3 grid(B / TPB), block(256);
    hipLaunchKernelGGL(moe_wmma_kernel, grid, block, 0, stream,
                       x, cp, W1, b1, W2, b2, w1t, pred, gts, eu, infl, cnt);
}
